// NodeModel_72215580115032
// MI455X (gfx1250) — matrix operand order repaired and checked by reading, not compiled, not run on hardware
//
#include <hip/hip_runtime.h>
#include <hip/hip_bf16.h>
#include <stdint.h>
#include <stddef.h>

// ---------------------------------------------------------------------------
// Types for CDNA5 WMMA (wave32): V_WMMA_F32_16X16X32_BF16
// ---------------------------------------------------------------------------
typedef __bf16 bf16_t;
typedef __attribute__((ext_vector_type(16))) __bf16 v16bf;
typedef __attribute__((ext_vector_type(8)))  float  v8f;

#define WMMA_BF16(A, B, C) \
  __builtin_amdgcn_wmma_f32_16x16x32_bf16(false, (A), false, (B), (short)0, (C), false, false)

__device__ __forceinline__ bf16_t f2bf(float f) {
  union { float f; uint32_t u; } v; v.f = f;
  uint32_t r = v.u + 0x7FFFu + ((v.u >> 16) & 1u);   // round-to-nearest-even
  uint16_t h = (uint16_t)(r >> 16);
  bf16_t o; __builtin_memcpy(&o, &h, 2); return o;
}

// A fragment: 16x32 bf16, ISA 7.12.2 layout. xp already includes the lane's
// K-half offset: halves 0..7 from xp[0..7], halves 8..15 from xp[16..23].
__device__ __forceinline__ void load_a(v16bf& a, const bf16_t* __restrict__ xp) {
  #pragma unroll
  for (int i = 0; i < 8; ++i) { a[i] = xp[i]; a[i + 8] = xp[i + 16]; }
}
// B fragment: 32x16 bf16; lane = N column, K rows 0..7 from wp rows 0..7, 8..15 from rows 16..23.
__device__ __forceinline__ void load_b(v16bf& b, const bf16_t* __restrict__ wp, size_t ld) {
  #pragma unroll
  for (int i = 0; i < 8; ++i) { b[i] = wp[(size_t)i * ld]; b[i + 8] = wp[(size_t)(i + 16) * ld]; }
}

// ---------------------------------------------------------------------------
// f32 -> bf16 conversion
// ---------------------------------------------------------------------------
__global__ void cvt_f32_bf16(const float* __restrict__ in, bf16_t* __restrict__ out,
                             long long n) {
  long long i = (long long)blockIdx.x * blockDim.x + threadIdx.x;
  if (i < n) out[i] = f2bf(in[i]);
}

// ---------------------------------------------------------------------------
// GEMM: Y[M,N] = relu( concat_K(X0[idx0],X1[idx1],X2[idx2]) @ W[Ktot,N] + bias )
// One wave per 16x64 tile; 4 x v_wmma_f32_16x16x32_bf16 per K-step of 32,
// software-pipelined 2 deep (ping-pong fragment registers) so next-step VMEM
// loads are in flight while the matrix pipe works on the current step.
// Requirements: M%16==0, N%64==0, each part K%32==0 (M=N_LAB=5000 breaks M%16==0).
// ---------------------------------------------------------------------------
__global__ __launch_bounds__(32)
void gemm_bf16_wmma(const bf16_t* __restrict__ X0, const int* __restrict__ i0, int K0,
                    const bf16_t* __restrict__ X1, const int* __restrict__ i1, int K1,
                    const bf16_t* __restrict__ X2, const int* __restrict__ i2, int K2,
                    const bf16_t* __restrict__ W,  const float* __restrict__ bias,
                    float* __restrict__ Y, int M, int N) {
  const int lane = threadIdx.x;            // wave32
  const int mb = blockIdx.x;               // 16-row tile
  const int nb = blockIdx.y;               // 64-col tile
  v8f acc[4] = {};

  const int arow  = mb * 16 + (lane & 15);      // A row for this lane
  const int khalf = (lane >> 4) << 3;           // 0 (lanes 0-15) or 8 (lanes 16-31)

  const bf16_t* Xs[3] = {X0, X1, X2};
  const int*    Is[3] = {i0, i1, i2};
  const int     Ks[3] = {K0, K1, K2};

  int wk = 0;  // running K offset into W rows (concat along K)
  for (int p = 0; p < 3; ++p) {
    const bf16_t* X = Xs[p];
    const int K = Ks[p];
    if (!X || K == 0) continue;
    const int* idx = Is[p];
    const int srow = idx ? idx[arow] : arow;
    const bf16_t* __restrict__ xp = X + (size_t)srow * (size_t)K + khalf;
    const bf16_t* __restrict__ wp =
        W + (size_t)(wk + khalf) * (size_t)N + (size_t)nb * 64 + (size_t)(lane & 15);
    const size_t wstep = (size_t)32 * (size_t)N;   // W rows per K-step
    const int nk = K >> 5;

    if ((nk & 1) == 0) {
      // --- pipelined path (nk even: K multiple of 64; always true here) ---
      v16bf a0, p0, p1, p2, p3;
      load_a(a0, xp);
      load_b(p0, wp, N); load_b(p1, wp + 16, N); load_b(p2, wp + 32, N); load_b(p3, wp + 48, N);
      for (int kk = 0; kk < nk; kk += 2) {
        v16bf a1, q0, q1, q2, q3;
        {
          const bf16_t* xq = xp + (size_t)(kk + 1) * 32;
          const bf16_t* wq = wp + (size_t)(kk + 1) * wstep;
          load_a(a1, xq);
          load_b(q0, wq, N); load_b(q1, wq + 16, N); load_b(q2, wq + 32, N); load_b(q3, wq + 48, N);
        }
        acc[0] = WMMA_BF16(a0, p0, acc[0]);
        acc[1] = WMMA_BF16(a0, p1, acc[1]);
        acc[2] = WMMA_BF16(a0, p2, acc[2]);
        acc[3] = WMMA_BF16(a0, p3, acc[3]);
        if (kk + 2 < nk) {
          const bf16_t* xr = xp + (size_t)(kk + 2) * 32;
          const bf16_t* wr = wp + (size_t)(kk + 2) * wstep;
          load_a(a0, xr);
          load_b(p0, wr, N); load_b(p1, wr + 16, N); load_b(p2, wr + 32, N); load_b(p3, wr + 48, N);
        }
        acc[0] = WMMA_BF16(a1, q0, acc[0]);
        acc[1] = WMMA_BF16(a1, q1, acc[1]);
        acc[2] = WMMA_BF16(a1, q2, acc[2]);
        acc[3] = WMMA_BF16(a1, q3, acc[3]);
      }
    } else {
      // --- generic fallback (any K%32==0) ---
      for (int kk = 0; kk < nk; ++kk) {
        v16bf a, b0, b1, b2, b3;
        const bf16_t* xq = xp + (size_t)kk * 32;
        const bf16_t* wq = wp + (size_t)kk * wstep;
        load_a(a, xq);
        load_b(b0, wq, N); load_b(b1, wq + 16, N); load_b(b2, wq + 32, N); load_b(b3, wq + 48, N);
        acc[0] = WMMA_BF16(a, b0, acc[0]);
        acc[1] = WMMA_BF16(a, b1, acc[1]);
        acc[2] = WMMA_BF16(a, b2, acc[2]);
        acc[3] = WMMA_BF16(a, b3, acc[3]);
      }
    }
    wk += K;
  }

  // Epilogue: bias + ReLU, store f32. C/D layout: VGPR r -> M=r+8*(lane>=16), N=lane&15.
  const int c = lane & 15;
  const int rbase = mb * 16 + ((lane >> 4) << 3);
  #pragma unroll
  for (int t = 0; t < 4; ++t) {
    const int col = nb * 64 + t * 16 + c;
    const float bv = bias[col];
    #pragma unroll
    for (int r = 0; r < 8; ++r) {
      float v = acc[t][r] + bv;
      v = v > 0.f ? v : 0.f;
      Y[(size_t)(rbase + r) * (size_t)N + col] = v;
    }
  }
}

// ---------------------------------------------------------------------------
// LayerNorm over last dim (N = 256 or 512), optionally fused with scatter-add
// (segment-sum numerator + per-row count) for the scatter_mean stages.
// One block (256 threads) per row.
// ---------------------------------------------------------------------------
__global__ __launch_bounds__(256)
void ln_kernel(const float* __restrict__ X, const float* __restrict__ g,
               const float* __restrict__ be, int N,
               float* __restrict__ outF, bf16_t* __restrict__ outB,
               float* __restrict__ ssum, float* __restrict__ scnt,
               const int* __restrict__ sidx) {
  const int row = blockIdx.x;
  const int tid = threadIdx.x;
  const int npt = N >> 8;  // 1 or 2 elems per thread
  const float* x = X + (size_t)row * (size_t)N;
  __shared__ float red[256];

  float v[2];
  float s = 0.f;
  for (int i = 0; i < npt; ++i) { v[i] = x[tid + (i << 8)]; s += v[i]; }
  red[tid] = s; __syncthreads();
  for (int o = 128; o > 0; o >>= 1) { if (tid < o) red[tid] += red[tid + o]; __syncthreads(); }
  const float mu = red[0] / (float)N;
  __syncthreads();

  s = 0.f;
  for (int i = 0; i < npt; ++i) { const float d = v[i] - mu; s += d * d; }
  red[tid] = s; __syncthreads();
  for (int o = 128; o > 0; o >>= 1) { if (tid < o) red[tid] += red[tid + o]; __syncthreads(); }
  const float inv = rsqrtf(red[0] / (float)N + 1e-5f);

  const int dst = ssum ? sidx[row] : 0;
  for (int i = 0; i < npt; ++i) {
    const int c = tid + (i << 8);
    const float y = (v[i] - mu) * inv * g[c] + be[c];
    if (outF) outF[(size_t)row * N + c] = y;
    if (outB) outB[(size_t)row * N + c] = f2bf(y);
    if (ssum) atomicAdd(&ssum[(size_t)dst * N + c], y);
  }
  if (ssum && tid == 0) atomicAdd(&scnt[dst], 1.0f);
}

// ---------------------------------------------------------------------------
// scatter_mean finalize: out_bf16 = sum / max(cnt,1)
// ---------------------------------------------------------------------------
__global__ void div_cvt_kernel(const float* __restrict__ sum, const float* __restrict__ cnt,
                               bf16_t* __restrict__ out, long long total, int N) {
  long long i = (long long)blockIdx.x * blockDim.x + threadIdx.x;
  if (i >= total) return;
  long long r = i / N;
  float c = cnt[r]; c = c < 1.f ? 1.f : c;
  out[i] = f2bf(sum[i] / c);
}

// ---------------------------------------------------------------------------
// Host side
// ---------------------------------------------------------------------------
namespace {

struct MlpPtrs { const float *w1, *b1, *g1, *be1, *w2, *b2, *g2, *be2; };

constexpr int N_INS = 50000, N_LAB = 5000;
constexpr int E_INS = 200000, E_LAB = 100000, E_CRS = 200000;
constexpr int LATENT = 512, FINAL = 256;

inline void launch_cvt(const float* in, bf16_t* out, long long n, hipStream_t s) {
  int blocks = (int)((n + 255) / 256);
  cvt_f32_bf16<<<blocks, 256, 0, s>>>(in, out, n);
}

void run_mlp(hipStream_t s,
             const bf16_t* X0, const int* i0, int K0,
             const bf16_t* X1, const int* i1, int K1,
             const bf16_t* X2, const int* i2, int K2,
             int M, const bf16_t* W1b, const bf16_t* W2b, const MlpPtrs& p,
             float* Hf, bf16_t* Hb, float* Zf,
             float* outF, bf16_t* outB,
             float* ssum, float* scnt, const int* sidx) {
  dim3 g1(M / 16, LATENT / 64);
  gemm_bf16_wmma<<<g1, 32, 0, s>>>(X0, i0, K0, X1, i1, K1, X2, i2, K2,
                                   W1b, p.b1, Hf, M, LATENT);
  ln_kernel<<<M, 256, 0, s>>>(Hf, p.g1, p.be1, LATENT, nullptr, Hb,
                              nullptr, nullptr, nullptr);
  dim3 g2(M / 16, FINAL / 64);
  gemm_bf16_wmma<<<g2, 32, 0, s>>>(Hb, nullptr, LATENT, nullptr, nullptr, 0,
                                   nullptr, nullptr, 0, W2b, p.b2, Zf, M, FINAL);
  ln_kernel<<<M, 256, 0, s>>>(Zf, p.g2, p.be2, FINAL, outF, outB, ssum, scnt, sidx);
}

} // namespace

extern "C" void kernel_launch(void* const* d_in, const int* in_sizes, int n_in,
                              void* d_out, int out_size, void* d_ws, size_t ws_size,
                              hipStream_t stream) {
  (void)n_in; (void)out_size; (void)ws_size;

  // ---- input index maps: insertion-order vs jax-sorted-key pytree flatten ----
  const bool sorted = (in_sizes[0] != N_INS * 256);  // insertion: node_ins first
  int ix_node_ins, ix_edge_ins, ix_node_lab, ix_edge_lab, ix_edge_crs;
  int ix_ei_ins, ix_ei_lab, ix_ei_crs, pstart;
  // logical MLP order: enc_ni enc_ei enc_nl enc_el enc_ec ins_inner ins_inter
  //                    ins_out lab_inner lab_inter lab_out
  static const int mlpSorted[11]  = {3, 1, 4, 2, 0, 5, 6, 7, 8, 9, 10};
  static const int leafSorted[8]  = {6, 0, 4, 2, 7, 1, 5, 3}; // w1,b1,g1,be1,w2,b2,g2,be2
  if (!sorted) {
    ix_node_ins = 0; ix_edge_ins = 1; ix_node_lab = 2; ix_edge_lab = 3; ix_edge_crs = 4;
    pstart = 5; ix_ei_ins = 93; ix_ei_lab = 94; ix_ei_crs = 95;
  } else {
    ix_edge_crs = 0; ix_edge_ins = 1; ix_edge_lab = 2;
    ix_ei_crs = 3; ix_ei_ins = 4; ix_ei_lab = 5;
    ix_node_ins = 6; ix_node_lab = 7; pstart = 8;
  }
  MlpPtrs mp[11];
  for (int l = 0; l < 11; ++l) {
    const int mbase = pstart + (sorted ? mlpSorted[l] : l) * 8;
    const float* leaf[8];
    for (int li = 0; li < 8; ++li)
      leaf[li] = (const float*)d_in[mbase + (sorted ? leafSorted[li] : li)];
    mp[l] = {leaf[0], leaf[1], leaf[2], leaf[3], leaf[4], leaf[5], leaf[6], leaf[7]};
  }
  static const int din[11] = {256, 128, 256, 128, 128, 512, 512, 768, 512, 512, 768};

  const float* node_ins = (const float*)d_in[ix_node_ins];
  const float* edge_ins = (const float*)d_in[ix_edge_ins];
  const float* node_lab = (const float*)d_in[ix_node_lab];
  const float* edge_lab = (const float*)d_in[ix_edge_lab];
  const float* edge_crs = (const float*)d_in[ix_edge_crs];
  const int* eii = (const int*)d_in[ix_ei_ins];
  const int* eil = (const int*)d_in[ix_ei_lab];
  const int* eic = (const int*)d_in[ix_ei_crs];
  const int *ri = eii, *ci = eii + E_INS;
  const int *rl = eil, *cl = eil + E_LAB;
  const int *rc = eic, *cc = eic + E_CRS;

  // ---- workspace bump allocator ----
  uint8_t* base = (uint8_t*)d_ws;
  size_t off = 0;
  auto alloc = [&](size_t bytes) -> void* {
    off = (off + 255) & ~(size_t)255;
    void* p = base + off; off += bytes; return p;
  };
  auto allocB = [&](size_t elems) { return (bf16_t*)alloc(elems * 2); };
  auto allocF = [&](size_t elems) { return (float*)alloc(elems * 4); };

  bf16_t* Xb_ni = allocB((size_t)N_INS * 256);
  bf16_t* Xb_ei = allocB((size_t)E_INS * 128);
  bf16_t* Xb_nl = allocB((size_t)N_LAB * 256);
  bf16_t* Xb_el = allocB((size_t)E_LAB * 128);
  bf16_t* Xb_ec = allocB((size_t)E_CRS * 128);
  bf16_t* W1b[11]; bf16_t* W2b[11];
  for (int l = 0; l < 11; ++l) {
    W1b[l] = allocB((size_t)din[l] * LATENT);
    W2b[l] = allocB((size_t)LATENT * FINAL);
  }
  bf16_t* ni_b = allocB((size_t)N_INS * 256);
  bf16_t* nl_b = allocB((size_t)N_LAB * 256);
  bf16_t* ei_b = allocB((size_t)E_INS * 256);
  bf16_t* el_b = allocB((size_t)E_LAB * 256);
  bf16_t* ec_b = allocB((size_t)E_CRS * 256);
  bf16_t* inner_b  = allocB((size_t)N_INS * 256);
  bf16_t* inter_b  = allocB((size_t)N_INS * 256);
  bf16_t* linner_b = allocB((size_t)N_LAB * 256);
  bf16_t* linter_b = allocB((size_t)N_LAB * 256);
  bf16_t* niout_b  = allocB((size_t)N_INS * 256);
  float*  Hf = allocF((size_t)E_CRS * LATENT);   // hidden f32 (reused); Zf aliases it
  bf16_t* Hb = allocB((size_t)E_CRS * LATENT);   // hidden bf16 (reused)
  float*  Zf = Hf;                               // layer-2 pre-LN output (safe alias)
  float*  sumB = allocF((size_t)N_INS * 256);    // scatter numerator (reused)
  float*  cntB = allocF((size_t)N_INS);          // scatter counts (reused)

  float* out    = (float*)d_out;
  float* out_ni = out;                // [50000,256]
  float* out_nl = out + 12800000;     // [5000,256]
  float* out_ei = out + 14080000;     // [200000,256]
  float* out_el = out + 65280000;     // [100000,256]
  float* out_ec = out + 90880000;     // [200000,256]

  // ---- convert inputs & weights to bf16 ----
  launch_cvt(node_ins, Xb_ni, (long long)N_INS * 256, stream);
  launch_cvt(edge_ins, Xb_ei, (long long)E_INS * 128, stream);
  launch_cvt(node_lab, Xb_nl, (long long)N_LAB * 256, stream);
  launch_cvt(edge_lab, Xb_el, (long long)E_LAB * 128, stream);
  launch_cvt(edge_crs, Xb_ec, (long long)E_CRS * 128, stream);
  for (int l = 0; l < 11; ++l) {
    launch_cvt(mp[l].w1, W1b[l], (long long)din[l] * LATENT, stream);
    launch_cvt(mp[l].w2, W2b[l], (long long)LATENT * FINAL, stream);
  }

  // ---- encoders ----
  run_mlp(stream, Xb_ni, nullptr, 256, nullptr, nullptr, 0, nullptr, nullptr, 0,
          N_INS, W1b[0], W2b[0], mp[0], Hf, Hb, Zf, nullptr, ni_b,
          nullptr, nullptr, nullptr);
  run_mlp(stream, Xb_ei, nullptr, 128, nullptr, nullptr, 0, nullptr, nullptr, 0,
          E_INS, W1b[1], W2b[1], mp[1], Hf, Hb, Zf, out_ei, ei_b,
          nullptr, nullptr, nullptr);
  run_mlp(stream, Xb_nl, nullptr, 256, nullptr, nullptr, 0, nullptr, nullptr, 0,
          N_LAB, W1b[2], W2b[2], mp[2], Hf, Hb, Zf, nullptr, nl_b,
          nullptr, nullptr, nullptr);
  run_mlp(stream, Xb_el, nullptr, 128, nullptr, nullptr, 0, nullptr, nullptr, 0,
          E_LAB, W1b[3], W2b[3], mp[3], Hf, Hb, Zf, out_el, el_b,
          nullptr, nullptr, nullptr);
  run_mlp(stream, Xb_ec, nullptr, 128, nullptr, nullptr, 0, nullptr, nullptr, 0,
          E_CRS, W1b[4], W2b[4], mp[4], Hf, Hb, Zf, out_ec, ec_b,
          nullptr, nullptr, nullptr);

  // ---- ins_inner: MLP(concat[ni[ri], ei]) -> scatter_mean by ci over N_INS ----
  hipMemsetAsync(sumB, 0, (size_t)N_INS * 256 * 4, stream);
  hipMemsetAsync(cntB, 0, (size_t)N_INS * 4, stream);
  run_mlp(stream, ni_b, ri, 256, ei_b, nullptr, 256, nullptr, nullptr, 0,
          E_INS, W1b[5], W2b[5], mp[5], Hf, Hb, Zf, nullptr, nullptr,
          sumB, cntB, ci);
  div_cvt_kernel<<<(int)(((long long)N_INS * 256 + 255) / 256), 256, 0, stream>>>(
      sumB, cntB, inner_b, (long long)N_INS * 256, 256);

  // ---- ins_inter: MLP(concat[nl[cc], ec]) -> scatter_mean by rc over N_INS ----
  hipMemsetAsync(sumB, 0, (size_t)N_INS * 256 * 4, stream);
  hipMemsetAsync(cntB, 0, (size_t)N_INS * 4, stream);
  run_mlp(stream, nl_b, cc, 256, ec_b, nullptr, 256, nullptr, nullptr, 0,
          E_CRS, W1b[6], W2b[6], mp[6], Hf, Hb, Zf, nullptr, nullptr,
          sumB, cntB, rc);
  div_cvt_kernel<<<(int)(((long long)N_INS * 256 + 255) / 256), 256, 0, stream>>>(
      sumB, cntB, inter_b, (long long)N_INS * 256, 256);

  // ---- ins_out: MLP(concat[ni, inner, inter]) -> ni_out ----
  run_mlp(stream, ni_b, nullptr, 256, inner_b, nullptr, 256, inter_b, nullptr, 256,
          N_INS, W1b[7], W2b[7], mp[7], Hf, Hb, Zf, out_ni, niout_b,
          nullptr, nullptr, nullptr);

  // ---- lab_inner: MLP(concat[nl[rl], el]) -> scatter_mean by cl over N_LAB ----
  hipMemsetAsync(sumB, 0, (size_t)N_LAB * 256 * 4, stream);
  hipMemsetAsync(cntB, 0, (size_t)N_LAB * 4, stream);
  run_mlp(stream, nl_b, rl, 256, el_b, nullptr, 256, nullptr, nullptr, 0,
          E_LAB, W1b[8], W2b[8], mp[8], Hf, Hb, Zf, nullptr, nullptr,
          sumB, cntB, cl);
  div_cvt_kernel<<<(int)(((long long)N_LAB * 256 + 255) / 256), 256, 0, stream>>>(
      sumB, cntB, linner_b, (long long)N_LAB * 256, 256);

  // ---- lab_inter: MLP(concat[ni_out[rc], ec]) -> scatter_mean by cc over N_LAB ----
  hipMemsetAsync(sumB, 0, (size_t)N_LAB * 256 * 4, stream);
  hipMemsetAsync(cntB, 0, (size_t)N_LAB * 4, stream);
  run_mlp(stream, niout_b, rc, 256, ec_b, nullptr, 256, nullptr, nullptr, 0,
          E_CRS, W1b[9], W2b[9], mp[9], Hf, Hb, Zf, nullptr, nullptr,
          sumB, cntB, cc);
  div_cvt_kernel<<<(int)(((long long)N_LAB * 256 + 255) / 256), 256, 0, stream>>>(
      sumB, cntB, linter_b, (long long)N_LAB * 256, 256);

  // ---- lab_out: MLP(concat[nl, linner, linter]) -> nl_out ----
  run_mlp(stream, nl_b, nullptr, 256, linner_b, nullptr, 256, linter_b, nullptr, 256,
          N_LAB, W1b[10], W2b[10], mp[10], Hf, Hb, Zf, out_nl, nullptr,
          nullptr, nullptr, nullptr);
}
